// TransformerEncoderLayer_52587579572816
// MI455X (gfx1250) — matrix operand order repaired and checked by reading, not compiled, not run on hardware
//
#include <hip/hip_runtime.h>

typedef _Float16 f16_t;
typedef __attribute__((ext_vector_type(16))) _Float16 v16h;
typedef __attribute__((ext_vector_type(8)))  _Float16 v8h;
typedef __attribute__((ext_vector_type(8)))  float    v8f;
typedef __attribute__((ext_vector_type(4)))  unsigned u32x4;
typedef __attribute__((ext_vector_type(8)))  int      i32x8;
typedef __attribute__((ext_vector_type(4)))  int      i32x4;

#define D_MODEL 768
#define NHEAD   12
#define DK      64
#define D_FFN   3072
#define SEQ     4096

#if defined(__has_builtin)
#  if __has_builtin(__builtin_amdgcn_tensor_load_to_lds)
#    define HAS_TDM 1
#  endif
#endif
#ifndef HAS_TDM
#  define HAS_TDM 0
#endif

// ---------------------------------------------------------------- fragment loaders
// A-matrix 16x32 f16 fragment (ISA 7.12.2): lane<16 holds M=lane, K=k0+[0..7],[16..23];
// lane>=16 holds M=lane-16, K=k0+[8..15],[24..31].
__device__ __forceinline__ v16h load_a_frag(const f16_t* base, int ld, int row0, int k0) {
  int lane = threadIdx.x & 31;
  int r = lane & 15, hi = lane >> 4;
  const f16_t* p = base + (size_t)(row0 + r) * ld + k0 + hi * 8;
  union { v16h v; v8h h[2]; } u;
  u.h[0] = *(const v8h*)(p);
  u.h[1] = *(const v8h*)(p + 16);
  return u.v;
}

// B-matrix 32x16 f16 fragment: lane<16 holds N=lane, K=k0+[0..7],[16..23]; lane>=16 K=k0+[8..15],[24..31].
// "W" stored row-major [N, K] => a B-column is a contiguous W row.
__device__ __forceinline__ v16h load_b_frag(const f16_t* base, int ld, int col0, int k0) {
  int lane = threadIdx.x & 31;
  int r = lane & 15, hi = lane >> 4;
  const f16_t* p = base + (size_t)(col0 + r) * ld + k0 + hi * 8;
  union { v16h v; v8h h[2]; } u;
  u.h[0] = *(const v8h*)(p);
  u.h[1] = *(const v8h*)(p + 16);
  return u.v;
}

__device__ __forceinline__ v8f wmma_f16(v16h a, v16h b, v8f c) {
  return __builtin_amdgcn_wmma_f32_16x16x32_f16(false, a, false, b, (short)0, c, false, false);
}

// ---------------------------------------------------------------- DPP 16-lane row reductions (no LDS)
template<int CTRL>
__device__ __forceinline__ float dpp_permf(float x) {
  return __builtin_bit_cast(float,
      __builtin_amdgcn_update_dpp(0, __builtin_bit_cast(int, x), CTRL, 0xF, 0xF, true));
}
__device__ __forceinline__ float row_max16(float x) {
  x = fmaxf(x, dpp_permf<0xB1>(x));   // quad_perm(1,0,3,2)  : xor 1
  x = fmaxf(x, dpp_permf<0x4E>(x));   // quad_perm(2,3,0,1)  : xor 2
  x = fmaxf(x, dpp_permf<0x141>(x));  // row_half_mirror     : merge groups of 8
  x = fmaxf(x, dpp_permf<0x140>(x));  // row_mirror          : merge halves of 16
  return x;
}
__device__ __forceinline__ float row_sum16(float x) {
  x += dpp_permf<0xB1>(x);
  x += dpp_permf<0x4E>(x);
  x += dpp_permf<0x141>(x);
  x += dpp_permf<0x140>(x);
  return x;
}

// ---------------------------------------------------------------- TDM: load 128x32 f16 tile -> LDS
#if HAS_TDM
__device__ __forceinline__ void tdm_tile_128x32(const f16_t* g, int ld_elems, void* lds) {
  unsigned long long ga = (unsigned long long)(uintptr_t)g;
  u32x4 g0 = { 1u,                                   // count=1, user mode
               (unsigned)(uintptr_t)lds,             // lds_addr (low 32 bits of generic ptr)
               (unsigned)ga,                         // global_addr[31:0]
               ((unsigned)(ga >> 32) & 0x01FFFFFFu)  // global_addr[56:32]
                 | 0x80000000u };                    // type=2 ("image") at bits 127:126
  i32x8 g1 = { (int)(1u << 16),                      // data_size=2B; no multicast/pad/iterate
               (int)(32u << 16),                     // tensor_dim0 = 32 (tile-local bounds)
               (int)(128u << 16),                    // tensor_dim1 = 128
               (int)(32u << 16),                     // tile_dim0 = 32
               128,                                  // tile_dim1 = 128, tile_dim2 = 0
               ld_elems,                             // tensor_dim0_stride = row stride (elems)
               0, 0 };
  i32x4 zz4 = { 0, 0, 0, 0 };
  i32x8 zz8 = { 0, 0, 0, 0, 0, 0, 0, 0 };
  __builtin_amdgcn_tensor_load_to_lds(g0, g1, zz4, zz4, zz8, 0);
}
#endif

// ---------------------------------------------------------------- cvt fp32 -> f16
__global__ void cvt_f32_f16_kernel(const float* __restrict__ src, f16_t* __restrict__ dst, int n) {
  int stride = gridDim.x * blockDim.x;
  for (int i = blockIdx.x * blockDim.x + threadIdx.x; i < n; i += stride)
    dst[i] = (f16_t)src[i];
}

// ---------------------------------------------------------------- layernorm (row of 768) -> f16
__global__ __launch_bounds__(256)
void layernorm_f16_kernel(const float* __restrict__ x, const float* __restrict__ gamma,
                          const float* __restrict__ beta, f16_t* __restrict__ out) {
  __shared__ float red[256];
  int row = blockIdx.x;
  const float* xr = x + (size_t)row * D_MODEL;
  int t = threadIdx.x;
  float v0 = xr[t], v1 = xr[t + 256], v2 = xr[t + 512];
  red[t] = v0 + v1 + v2;
  __syncthreads();
  for (int off = 128; off > 0; off >>= 1) { if (t < off) red[t] += red[t + off]; __syncthreads(); }
  float mu = red[0] * (1.0f / D_MODEL);
  __syncthreads();
  float d0 = v0 - mu, d1 = v1 - mu, d2 = v2 - mu;
  red[t] = d0 * d0 + d1 * d1 + d2 * d2;
  __syncthreads();
  for (int off = 128; off > 0; off >>= 1) { if (t < off) red[t] += red[t + off]; __syncthreads(); }
  float rstd = rsqrtf(red[0] * (1.0f / D_MODEL) + 1e-5f);
  f16_t* o = out + (size_t)row * D_MODEL;
  o[t]       = (f16_t)(d0 * rstd * gamma[t]       + beta[t]);
  o[t + 256] = (f16_t)(d1 * rstd * gamma[t + 256] + beta[t + 256]);
  o[t + 512] = (f16_t)(d2 * rstd * gamma[t + 512] + beta[t + 512]);
}

// ---------------------------------------------------------------- V[s, h*64+d] -> Vt[h][d][s]
__global__ void transpose_v_kernel(const f16_t* __restrict__ V, f16_t* __restrict__ Vt) {
  int n = NHEAD * DK * SEQ;
  int stride = gridDim.x * blockDim.x;
  for (int i = blockIdx.x * blockDim.x + threadIdx.x; i < n; i += stride) {
    int s  = i & (SEQ - 1);
    int hd = i >> 12;                       // h*64 + d
    Vt[i] = V[(size_t)s * D_MODEL + hd];
  }
}

// ---------------------------------------------------------------- GEMM: out[M,N] = A[M,K] @ W[N,K]^T + bias
// 256 threads = 8 waves (4m x 2n); block tile 128x128; wave tile 32x64; K-step 32.
// TDM double-buffers 128x32 A/B tiles into LDS while WMMAs consume the other buffer.
template<bool RELU, bool RESID, bool OUT_F16>
__global__ __launch_bounds__(256)
void gemm_wmma_kernel(const f16_t* __restrict__ A, const f16_t* __restrict__ W,
                      const float* __restrict__ bias, const float* __restrict__ resid,
                      float* __restrict__ outF, f16_t* __restrict__ outB,
                      int M, int N, int K) {
#if HAS_TDM
  __shared__ f16_t Atile[2][128 * 32];
  __shared__ f16_t Btile[2][128 * 32];
#else
  __shared__ f16_t Atile[1][128 * 32];
  __shared__ f16_t Btile[1][128 * 32];
#endif
  int wave = threadIdx.x >> 5;
  int lane = threadIdx.x & 31;
  int wm = wave & 3, wn = wave >> 2;
  int rowB = blockIdx.y * 128;              // block base (global)
  int colB = blockIdx.x * 128;
  int row0 = wm * 32;                       // wave base (block-relative)
  int col0 = wn * 64;

  const v8f zero = {0.f, 0.f, 0.f, 0.f, 0.f, 0.f, 0.f, 0.f};
  v8f acc[2][4];
  #pragma unroll
  for (int i = 0; i < 2; i++)
    #pragma unroll
    for (int j = 0; j < 4; j++) acc[i][j] = zero;

  const int nk = K >> 5;

#if HAS_TDM
  if (wave == 0) {
    tdm_tile_128x32(A + (size_t)rowB * K, K, &Atile[0][0]);
    tdm_tile_128x32(W + (size_t)colB * K, K, &Btile[0][0]);
  }
  for (int i = 0; i < nk; i++) {
    int cur = i & 1;
    if (i + 1 < nk) {
      if (wave == 0) {
        int k0 = (i + 1) << 5;
        tdm_tile_128x32(A + (size_t)rowB * K + k0, K, &Atile[1 - cur][0]);
        tdm_tile_128x32(W + (size_t)colB * K + k0, K, &Btile[1 - cur][0]);
      }
      __builtin_amdgcn_s_wait_tensorcnt(2);   // current buffer's 2 loads complete
    } else {
      __builtin_amdgcn_s_wait_tensorcnt(0);
    }
    __syncthreads();                          // LDS tiles visible to all waves
    v16h a0 = load_a_frag(&Atile[cur][0], 32, row0,      0);
    v16h a1 = load_a_frag(&Atile[cur][0], 32, row0 + 16, 0);
    #pragma unroll
    for (int j = 0; j < 4; j++) {
      v16h b = load_b_frag(&Btile[cur][0], 32, col0 + 16 * j, 0);
      acc[0][j] = wmma_f16(a0, b, acc[0][j]);
      acc[1][j] = wmma_f16(a1, b, acc[1][j]);
    }
    __syncthreads();                          // all reads done before TDM overwrites this buffer
  }
#else
  for (int i = 0; i < nk; i++) {
    int k0 = i << 5;
    __syncthreads();
    for (int c = threadIdx.x; c < 512; c += 256) {
      int rr = c >> 2, cc = (c & 3) * 8;
      *(v8h*)&Atile[0][rr * 32 + cc] = *(const v8h*)(A + (size_t)(rowB + rr) * K + k0 + cc);
      *(v8h*)&Btile[0][rr * 32 + cc] = *(const v8h*)(W + (size_t)(colB + rr) * K + k0 + cc);
    }
    __syncthreads();
    v16h a0 = load_a_frag(&Atile[0][0], 32, row0,      0);
    v16h a1 = load_a_frag(&Atile[0][0], 32, row0 + 16, 0);
    #pragma unroll
    for (int j = 0; j < 4; j++) {
      v16h b = load_b_frag(&Btile[0][0], 32, col0 + 16 * j, 0);
      acc[0][j] = wmma_f16(a0, b, acc[0][j]);
      acc[1][j] = wmma_f16(a1, b, acc[1][j]);
    }
  }
#endif

  int r = lane & 15, hi = lane >> 4;
  #pragma unroll
  for (int j = 0; j < 4; j++) {
    int col = colB + col0 + 16 * j + r;
    float bcol = bias[col];
    #pragma unroll
    for (int i = 0; i < 2; i++) {
      int rowBase = rowB + row0 + i * 16 + hi * 8;
      #pragma unroll
      for (int v = 0; v < 8; v++) {
        float val = acc[i][j][v] + bcol;
        if (RELU) val = fmaxf(val, 0.0f);
        size_t idx = (size_t)(rowBase + v) * N + col;
        if (RESID) val += resid[idx];
        if (OUT_F16) outB[idx] = (f16_t)val; else outF[idx] = val;
      }
    }
  }
}

// ---------------------------------------------------------------- flash attention (one head, 64 queries / block)
// 128 threads = 4 waves, each owns a 16-query tile; online softmax == exp(s-max)/(sum+1e-10).
__global__ __launch_bounds__(128)
void flash_attn_kernel(const f16_t* __restrict__ Q, const f16_t* __restrict__ Km,
                       const f16_t* __restrict__ Vt, f16_t* __restrict__ O) {
  __shared__ f16_t plds[4][16 * 32];
  int wave = threadIdx.x >> 5;
  int lane = threadIdx.x & 31;
  int r = lane & 15, hi = lane >> 4;
  int h  = blockIdx.y;
  int q0 = blockIdx.x * 64 + wave * 16;
  const float scale = 0.125f;                       // 1/sqrt(64)

  v16h qa0 = load_a_frag(Q + h * DK, D_MODEL, q0, 0);
  v16h qa1 = load_a_frag(Q + h * DK, D_MODEL, q0, 32);

  const v8f zero = {0.f, 0.f, 0.f, 0.f, 0.f, 0.f, 0.f, 0.f};
  v8f acc[4];
  #pragma unroll
  for (int t = 0; t < 4; t++) acc[t] = zero;
  float mrow[8], lrow[8];
  #pragma unroll
  for (int v = 0; v < 8; v++) { mrow[v] = -1e30f; lrow[v] = 0.0f; }

  f16_t* pl = plds[wave];
  const f16_t* vtb = Vt + (size_t)h * DK * SEQ;

  for (int kb = 0; kb < SEQ; kb += 32) {
    // scores: 16 queries x 32 keys, d=64 split into two K=32 WMMAs
    v8f sc[2];
    #pragma unroll
    for (int j = 0; j < 2; j++) {
      v16h kb0 = load_b_frag(Km + h * DK, D_MODEL, kb + 16 * j, 0);
      v16h kb1 = load_b_frag(Km + h * DK, D_MODEL, kb + 16 * j, 32);
      v8f s = zero;
      s = wmma_f16(qa0, kb0, s);
      s = wmma_f16(qa1, kb1, s);
      sc[j] = s;
    }
    // online softmax update (frag row = v + hi*8; columns live across the 16-lane half)
    float p0[8], p1[8], alpha[8];
    #pragma unroll
    for (int v = 0; v < 8; v++) {
      p0[v] = sc[0][v] * scale;
      p1[v] = sc[1][v] * scale;
      float bm = row_max16(fmaxf(p0[v], p1[v]));
      float mnew = fmaxf(mrow[v], bm);
      alpha[v] = __expf(mrow[v] - mnew);
      mrow[v] = mnew;
      p0[v] = __expf(p0[v] - mnew);
      p1[v] = __expf(p1[v] - mnew);
      lrow[v] = lrow[v] * alpha[v] + row_sum16(p0[v] + p1[v]);
    }
    #pragma unroll
    for (int t = 0; t < 4; t++)
      #pragma unroll
      for (int v = 0; v < 8; v++) acc[t][v] *= alpha[v];

    // C-frag layout -> A-frag layout via per-wave LDS bounce
    #pragma unroll
    for (int v = 0; v < 8; v++) {
      pl[(v + hi * 8) * 32 + r]      = (f16_t)p0[v];
      pl[(v + hi * 8) * 32 + 16 + r] = (f16_t)p1[v];
    }
    v16h pa;
    {
      union { v16h v; v8h h2[2]; } u;
      const f16_t* pp = pl + r * 32 + hi * 8;
      u.h2[0] = *(const v8h*)(pp);
      u.h2[1] = *(const v8h*)(pp + 16);
      pa = u.v;
    }
    // O += P @ V : B columns are contiguous rows of Vt[h][d][:]
    #pragma unroll
    for (int t = 0; t < 4; t++) {
      v16h bv = load_b_frag(vtb, SEQ, t * 16, kb);
      acc[t] = wmma_f16(pa, bv, acc[t]);
    }
  }

  #pragma unroll
  for (int v = 0; v < 8; v++) {
    float inv = 1.0f / (lrow[v] + 1e-10f);
    int row = q0 + v + hi * 8;
    #pragma unroll
    for (int t = 0; t < 4; t++)
      O[(size_t)row * D_MODEL + h * DK + t * 16 + r] = (f16_t)(acc[t][v] * inv);
  }
}

// ---------------------------------------------------------------- launch
static inline void* ws_take(char*& p, size_t bytes) {
  void* r = (void*)p;
  p += (bytes + 255) & ~(size_t)255;
  return r;
}

extern "C" void kernel_launch(void* const* d_in, const int* in_sizes, int n_in,
                              void* d_out, int out_size, void* d_ws, size_t ws_size,
                              hipStream_t stream) {
  (void)in_sizes; (void)n_in; (void)out_size; (void)ws_size;
  const float* x   = (const float*)d_in[0];
  const float* Wq  = (const float*)d_in[1];  const float* bq = (const float*)d_in[2];
  const float* Wk  = (const float*)d_in[3];  const float* bk = (const float*)d_in[4];
  const float* Wv  = (const float*)d_in[5];  const float* bv = (const float*)d_in[6];
  const float* Wo  = (const float*)d_in[7];  const float* bo = (const float*)d_in[8];
  const float* W1  = (const float*)d_in[9];  const float* b1 = (const float*)d_in[10];
  const float* W2  = (const float*)d_in[11]; const float* b2 = (const float*)d_in[12];
  const float* g1  = (const float*)d_in[13]; const float* be1 = (const float*)d_in[14];
  const float* g2  = (const float*)d_in[15]; const float* be2 = (const float*)d_in[16];

  char* p = (char*)d_ws;
  const size_t DD = (size_t)D_MODEL * D_MODEL;
  const size_t DF = (size_t)D_FFN * D_MODEL;
  const size_t SD = (size_t)SEQ * D_MODEL;
  const size_t SF = (size_t)SEQ * D_FFN;

  f16_t* wq16 = (f16_t*)ws_take(p, DD * 2);
  f16_t* wk16 = (f16_t*)ws_take(p, DD * 2);
  f16_t* wv16 = (f16_t*)ws_take(p, DD * 2);
  f16_t* wo16 = (f16_t*)ws_take(p, DD * 2);
  f16_t* w116 = (f16_t*)ws_take(p, DF * 2);
  f16_t* w216 = (f16_t*)ws_take(p, DF * 2);
  f16_t* xn16 = (f16_t*)ws_take(p, SD * 2);
  f16_t* q16  = (f16_t*)ws_take(p, SD * 2);
  f16_t* k16  = (f16_t*)ws_take(p, SD * 2);
  f16_t* v16_ = (f16_t*)ws_take(p, SD * 2);
  f16_t* vt16 = (f16_t*)ws_take(p, SD * 2);
  f16_t* at16 = (f16_t*)ws_take(p, SD * 2);
  float* x1f  = (float*) ws_take(p, SD * 4);
  f16_t* xn2  = (f16_t*)ws_take(p, SD * 2);
  f16_t* h16  = (f16_t*)ws_take(p, SF * 2);

  // weights -> f16
  cvt_f32_f16_kernel<<<1024, 256, 0, stream>>>(Wq, wq16, (int)DD);
  cvt_f32_f16_kernel<<<1024, 256, 0, stream>>>(Wk, wk16, (int)DD);
  cvt_f32_f16_kernel<<<1024, 256, 0, stream>>>(Wv, wv16, (int)DD);
  cvt_f32_f16_kernel<<<1024, 256, 0, stream>>>(Wo, wo16, (int)DD);
  cvt_f32_f16_kernel<<<1024, 256, 0, stream>>>(W1, w116, (int)DF);
  cvt_f32_f16_kernel<<<1024, 256, 0, stream>>>(W2, w216, (int)DF);

  // sublayer 1: LN -> QKV -> attention -> Wo + residual
  layernorm_f16_kernel<<<SEQ, 256, 0, stream>>>(x, g1, be1, xn16);

  dim3 gDD(D_MODEL / 128, SEQ / 128);               // (6, 32)
  gemm_wmma_kernel<false, false, true><<<gDD, 256, 0, stream>>>(
      xn16, wq16, bq, nullptr, nullptr, q16, SEQ, D_MODEL, D_MODEL);
  gemm_wmma_kernel<false, false, true><<<gDD, 256, 0, stream>>>(
      xn16, wk16, bk, nullptr, nullptr, k16, SEQ, D_MODEL, D_MODEL);
  gemm_wmma_kernel<false, false, true><<<gDD, 256, 0, stream>>>(
      xn16, wv16, bv, nullptr, nullptr, v16_, SEQ, D_MODEL, D_MODEL);

  transpose_v_kernel<<<1024, 256, 0, stream>>>(v16_, vt16);

  flash_attn_kernel<<<dim3(SEQ / 64, NHEAD), 128, 0, stream>>>(q16, k16, vt16, at16);

  gemm_wmma_kernel<false, true, false><<<gDD, 256, 0, stream>>>(
      at16, wo16, bo, x, x1f, nullptr, SEQ, D_MODEL, D_MODEL);

  // sublayer 2: LN -> FFN + residual
  layernorm_f16_kernel<<<SEQ, 256, 0, stream>>>(x1f, g2, be2, xn2);

  dim3 gDF(D_FFN / 128, SEQ / 128);                 // (24, 32)
  gemm_wmma_kernel<true, false, true><<<gDF, 256, 0, stream>>>(
      xn2, w116, b1, nullptr, nullptr, h16, SEQ, D_FFN, D_MODEL);

  gemm_wmma_kernel<false, true, false><<<gDD, 256, 0, stream>>>(
      h16, w216, b2, x1f, (float*)d_out, nullptr, SEQ, D_MODEL, D_FFN);
}
